// Mask2FormerPixelDecoder_27144193311282
// MI455X (gfx1250) — compile-verified, matrix operand order checked by reading, not run on hardware
//
#include <hip/hip_runtime.h>
#include <hip/hip_bf16.h>

// ---------------- static config (matches reference) ----------------
#define LEN   18720
#define CDIM  256
#define HEADS 8
#define CH    32          // CDIM / HEADS  == wave32 lanes, one lane per channel
#define NLVL  4
#define NPTS  4
#define FF    1024
#define NL    6

typedef __attribute__((ext_vector_type(16))) _Float16 v16h;
typedef __attribute__((ext_vector_type(8)))  _Float16 v8h;
typedef __attribute__((ext_vector_type(8)))  float    v8f;

// level geometry: (D,H,W) = (16,32,32),(8,16,16),(4,8,8),(2,4,4)
__device__ __forceinline__ void token_level(int t, int& lt, int& it) {
    if      (t < 16384) { lt = 0; it = t;         }
    else if (t < 18432) { lt = 1; it = t - 16384; }
    else if (t < 18688) { lt = 2; it = t - 18432; }
    else                { lt = 3; it = t - 18688; }
}

// ---------------- weight pack: f32 [K,N] -> f16 WMMA B-fragments --------------
// Fragment layout (per 16-col tile nt, per 32-K step kk, per lane):
//   16 contiguous halves; half e corresponds to
//   k = kk*32 + (lane>>4)*8 + (e&7) + (e>>3)*16 , col = nt*16 + (lane&15)
__global__ void pack_b_kernel(const float* __restrict__ B, _Float16* __restrict__ Bp,
                              int N, int K)
{
    const int id    = blockIdx.x * blockDim.x + threadIdx.x;
    const int total = (N / 16) * (K / 32) * 32;
    if (id >= total) return;
    const int lane = id & 31;
    const int kk   = (id >> 5) % (K / 32);
    const int nt   = (id >> 5) / (K / 32);
    const int l15  = lane & 15;
    const int g    = lane >> 4;
    _Float16* __restrict__ dst = Bp + (size_t)id * 16;
    const float* __restrict__ bcol = B + nt * 16 + l15;
    const int kb = kk * 32 + g * 8;
    #pragma unroll
    for (int e = 0; e < 8; ++e) {
        dst[e]     = (_Float16)bcol[(size_t)(kb + e)      * N];
        dst[e + 8] = (_Float16)bcol[(size_t)(kb + 16 + e) * N];
    }
}

__device__ __forceinline__ v16h load_a_frag(const _Float16* __restrict__ arow, int kk) {
    const v8h alo = *(const v8h*)(arow + kk * 32);
    const v8h ahi = *(const v8h*)(arow + kk * 32 + 16);
    v16h a;
    #pragma unroll
    for (int e = 0; e < 8; ++e) { a[e] = alo[e]; a[e + 8] = ahi[e]; }
    return a;
}

// ---------------- WMMA GEMM: Out[M,N] = (relu?)(A[M,K] @ B + bias[N]) ---------
// A is f16 row-major (pre-converted by producers); B pre-packed f16 fragments.
// Each wave computes a 32x64 strip (2 M-tiles x 4 N-tiles): per K-step four
// 16B A loads and four contiguous 32B B loads feed 8 v_wmma (1.5 loads/wmma).
template<int K, bool RELU, bool OUTHALF>
__global__ __launch_bounds__(256)
void gemm_wmma_kernel(const _Float16* __restrict__ A, const _Float16* __restrict__ Bp,
                      const float* __restrict__ bias, void* __restrict__ OutV,
                      int M, int N)
{
    const int wave = threadIdx.x >> 5;
    const int lane = threadIdx.x & 31;
    const int m0 = blockIdx.y * 256 + wave * 32;
    const int n0 = blockIdx.x * 64;
    if (m0 >= M) return;                      // per-wave exit, no barriers used

    const int l15 = lane & 15;
    const int g   = lane >> 4;
    constexpr int KS = K / 32;                // K-steps

    v8f acc[2][4] = {};
    const _Float16* __restrict__ arow0 = A + (size_t)(m0 + l15) * K + g * 8;
    const _Float16* __restrict__ arow1 = arow0 + (size_t)16 * K;
    const v16h* __restrict__ bp0 = (const v16h*)(Bp) + ((size_t)(n0 / 16) * KS) * 32 + lane;

    for (int kk = 0; kk < KS; ++kk) {
        const v16h a0 = load_a_frag(arow0, kk);
        const v16h a1 = load_a_frag(arow1, kk);
        #pragma unroll
        for (int c = 0; c < 4; ++c) {
            const v16h b = bp0[(size_t)(c * KS + kk) * 32];
            acc[0][c] = __builtin_amdgcn_wmma_f32_16x16x32_f16(
                            false, a0, false, b, (short)0, acc[0][c], false, false);
            acc[1][c] = __builtin_amdgcn_wmma_f32_16x16x32_f16(
                            false, a1, false, b, (short)0, acc[1][c], false, false);
        }
    }

    // epilogue: C/D layout -> lane group g covers rows base+g*8+j, col n0+c*16+l15
    #pragma unroll
    for (int r = 0; r < 2; ++r) {
        #pragma unroll
        for (int c = 0; c < 4; ++c) {
            const int col = n0 + c * 16 + l15;
            const float bc = bias[col];
            #pragma unroll
            for (int j = 0; j < 8; ++j) {
                const int row = m0 + r * 16 + g * 8 + j;
                float v = acc[r][c][j] + bc;
                if (RELU) v = fmaxf(v, 0.0f);
                if (OUTHALF) ((_Float16*)OutV)[(size_t)row * N + col] = (_Float16)v;
                else         ((float*)   OutV)[(size_t)row * N + col] = v;
            }
        }
    }
}

// ---------------- init: h(f32) = x ; h16 = (f16)x -----------------------------
__global__ void copy_cvt_kernel(const float* __restrict__ x, float* __restrict__ h,
                                _Float16* __restrict__ h16)
{
    int i = blockIdx.x * blockDim.x + threadIdx.x;
    if (i >= LEN * CDIM) return;
    float v = x[i];
    h[i] = v;
    h16[i] = (_Float16)v;
}

// ---------------- elementwise: p = pos + level_embed[level(token)] ------------
__global__ void pos_level_kernel(const float* __restrict__ pos,
                                 const float* __restrict__ le,
                                 float* __restrict__ p)
{
    int i = blockIdx.x * blockDim.x + threadIdx.x;
    if (i >= LEN * CDIM) return;
    int t = i >> 8, c = i & 255;
    int lt, it; token_level(t, lt, it);
    p[i] = pos[i] + le[lt * CDIM + c];
}

// ---------------- q16 = (f16)(h + p) -- q only feeds GEMM A-operands ----------
__global__ void add_f16_kernel(const float* __restrict__ a, const float* __restrict__ b,
                               _Float16* __restrict__ o, int n)
{
    int i = blockIdx.x * blockDim.x + threadIdx.x;
    if (i < n) o[i] = (_Float16)(a[i] + b[i]);
}

// ---------------- softmax over 16 (L*P) per (token,head) ---------------------
__global__ void softmax16_kernel(float* __restrict__ aw, int rows)
{
    int r = blockIdx.x * blockDim.x + threadIdx.x;
    if (r >= rows) return;
    float* p = aw + (size_t)r * 16;
    float m = p[0], e[16];
    #pragma unroll
    for (int i = 1; i < 16; ++i) m = fmaxf(m, p[i]);
    float s = 0.0f;
    #pragma unroll
    for (int i = 0; i < 16; ++i) { e[i] = __expf(p[i] - m); s += e[i]; }
    float inv = 1.0f / s;
    #pragma unroll
    for (int i = 0; i < 16; ++i) p[i] = e[i] * inv;
}

// ---------------- deformable trilinear sampling ------------------------------
// one wave32 per (token, head); lane = channel -> coalesced 128B gathers.
// Output written in f16 (it only feeds the Wout GEMM A-operand).
__global__ __launch_bounds__(256)
void deform_sample_kernel(const float* __restrict__ value,  // [LEN, 256] f32
                          const float* __restrict__ off,    // [LEN, H*L*P*3]
                          const float* __restrict__ aw,     // [LEN, H*L*P]
                          _Float16* __restrict__ out)       // [LEN, 256] f16
{
    const int wid  = blockIdx.x * (blockDim.x >> 5) + (threadIdx.x >> 5);
    const int lane = threadIdx.x & 31;
    if (wid >= LEN * HEADS) return;
    const int t = wid >> 3;
    const int h = wid & 7;

    const int  Ds[4] = {16, 8, 4, 2};
    const int  Hs[4] = {32, 16, 8, 4};
    const int  Ws[4] = {32, 16, 8, 4};
    const int  St[4] = {0, 16384, 18432, 18688};

    int lt, it; token_level(t, lt, it);
    const int Wt = Ws[lt], Ht = Hs[lt];
    const int xt = it % Wt, yt = (it / Wt) % Ht, zt = it / (Wt * Ht);
    const float rx = (xt + 0.5f) / Wt;
    const float ry = (yt + 0.5f) / Ht;
    const float rz = (zt + 0.5f) / Ds[lt];

    const float* __restrict__ offp = off + (size_t)t * (HEADS * NLVL * NPTS * 3)
                                         + h * (NLVL * NPTS * 3);
    const float* __restrict__ awp  = aw  + (size_t)t * (HEADS * NLVL * NPTS)
                                         + h * (NLVL * NPTS);
    float acc = 0.0f;

    #pragma unroll
    for (int lvl = 0; lvl < NLVL; ++lvl) {
        const int W = Ws[lvl], H = Hs[lvl], D = Ds[lvl];
        const float* __restrict__ vb =
            value + (size_t)St[lvl] * CDIM + h * CH + lane;
        #pragma unroll
        for (int p = 0; p < NPTS; ++p) {
            const float ox = offp[(lvl * NPTS + p) * 3 + 0];
            const float oy = offp[(lvl * NPTS + p) * 3 + 1];
            const float oz = offp[(lvl * NPTS + p) * 3 + 2];
            const float w  = awp[lvl * NPTS + p];
            const float gx = rx * W + ox - 0.5f;
            const float gy = ry * H + oy - 0.5f;
            const float gz = rz * D + oz - 0.5f;
            const float fx0 = floorf(gx), fy0 = floorf(gy), fz0 = floorf(gz);
            const float fx = gx - fx0, fy = gy - fy0, fz = gz - fz0;
            const int x0 = (int)fx0, y0 = (int)fy0, z0 = (int)fz0;
            float s = 0.0f;
            #pragma unroll
            for (int dz = 0; dz < 2; ++dz)
            #pragma unroll
            for (int dy = 0; dy < 2; ++dy)
            #pragma unroll
            for (int dx = 0; dx < 2; ++dx) {
                const int xi = x0 + dx, yi = y0 + dy, zi = z0 + dz;
                const float cw = (dx ? fx : 1.0f - fx) *
                                 (dy ? fy : 1.0f - fy) *
                                 (dz ? fz : 1.0f - fz);
                const bool valid = (xi >= 0) & (xi < W) & (yi >= 0) & (yi < H) &
                                   (zi >= 0) & (zi < D);
                const int xc = min(max(xi, 0), W - 1);
                const int yc = min(max(yi, 0), H - 1);
                const int zc = min(max(zi, 0), D - 1);
                const int ci = zc * (H * W) + yc * W + xc;
                const float g = vb[(size_t)ci * CDIM];   // coalesced across lanes
                s += valid ? cw * g : 0.0f;
            }
            acc += w * s;
        }
    }
    out[(size_t)t * CDIM + h * CH + lane] = (_Float16)acc;
}

// ---------------- fused residual + LayerNorm, one wave32 per token -----------
// Writes f32 result and (optionally) an f16 shadow for downstream GEMM A use.
__global__ __launch_bounds__(256)
void add_ln_kernel(const float* __restrict__ hin, const float* __restrict__ add,
                   const float* __restrict__ s, const float* __restrict__ b,
                   float* __restrict__ hout, _Float16* __restrict__ hout16)
{
    const int wid  = blockIdx.x * (blockDim.x >> 5) + (threadIdx.x >> 5);
    const int lane = threadIdx.x & 31;
    if (wid >= LEN) return;
    const float* __restrict__ hp = hin + (size_t)wid * CDIM;
    const float* __restrict__ ap = add + (size_t)wid * CDIM;
    float v[8];
    float sum = 0.0f;
    #pragma unroll
    for (int i = 0; i < 8; ++i) {
        v[i] = hp[lane + 32 * i] + ap[lane + 32 * i];
        sum += v[i];
    }
    #pragma unroll
    for (int o = 16; o > 0; o >>= 1) sum += __shfl_xor(sum, o, 32);
    const float mean = sum * (1.0f / CDIM);
    float var = 0.0f;
    #pragma unroll
    for (int i = 0; i < 8; ++i) { float d = v[i] - mean; var += d * d; }
    #pragma unroll
    for (int o = 16; o > 0; o >>= 1) var += __shfl_xor(var, o, 32);
    const float inv = rsqrtf(var * (1.0f / CDIM) + 1e-5f);
    float* __restrict__ op = hout + (size_t)wid * CDIM;
    #pragma unroll
    for (int i = 0; i < 8; ++i) {
        const int c = lane + 32 * i;
        const float r = (v[i] - mean) * inv * s[c] + b[c];
        op[c] = r;
        if (hout16) hout16[(size_t)wid * CDIM + c] = (_Float16)r;
    }
}

// ---------------- host orchestration -----------------------------------------
extern "C" void kernel_launch(void* const* d_in, const int* in_sizes, int n_in,
                              void* d_out, int out_size, void* d_ws, size_t ws_size,
                              hipStream_t stream)
{
    (void)in_sizes; (void)n_in; (void)out_size; (void)ws_size;
    const float* x    = (const float*)d_in[0];
    const float* pos  = (const float*)d_in[1];
    const float* le   = (const float*)d_in[2];
    const float* Wv   = (const float*)d_in[3];
    const float* bv   = (const float*)d_in[4];
    const float* Woff = (const float*)d_in[5];
    const float* boff = (const float*)d_in[6];
    const float* Wa   = (const float*)d_in[7];
    const float* ba   = (const float*)d_in[8];
    const float* Wout = (const float*)d_in[9];
    const float* bout = (const float*)d_in[10];
    const float* ln1s = (const float*)d_in[11];
    const float* ln1b = (const float*)d_in[12];
    const float* W1   = (const float*)d_in[13];
    const float* b1   = (const float*)d_in[14];
    const float* W2   = (const float*)d_in[15];
    const float* b2   = (const float*)d_in[16];
    const float* ln2s = (const float*)d_in[17];
    const float* ln2b = (const float*)d_in[18];
    float* out = (float*)d_out;

    // ---------------- workspace layout ----------------
    float* ws   = (float*)d_ws;
    float* hbuf = ws;                                  // LEN*256  f32 (residual)
    float* pbuf = hbuf + (size_t)LEN * CDIM;           // LEN*256  f32
    float* tmp  = pbuf + (size_t)LEN * CDIM;           // LEN*256  f32
    float* vbuf = tmp  + (size_t)LEN * CDIM;           // LEN*256  f32 (value)
    float* obuf = vbuf + (size_t)LEN * CDIM;           // LEN*384  f32 (offsets)
    float* abuf = obuf + (size_t)LEN * 384;            // LEN*128  f32 (attn w)
    _Float16* H16   = (_Float16*)(abuf + (size_t)LEN * 128);
    _Float16* h16   = H16;                             // LEN*256  f16 shadow of h
    _Float16* q16   = h16  + (size_t)LEN * CDIM;       // LEN*256  f16
    _Float16* at16  = q16  + (size_t)LEN * CDIM;       // LEN*256  f16 (attnout)
    _Float16* ff16  = at16 + (size_t)LEN * CDIM;       // LEN*1024 f16 (FF mid)
    _Float16* Pk    = ff16 + (size_t)LEN * FF;         // packed weights (f16)
    _Float16* Wv_p   = Pk;                             // 256*256
    _Float16* Woff_p = Wv_p   + 256 * 256;             // 256*384
    _Float16* Wa_p   = Woff_p + 256 * 384;             // 256*128
    _Float16* Wout_p = Wa_p   + 256 * 128;             // 256*256
    _Float16* W1_p   = Wout_p + 256 * 256;             // 256*1024
    _Float16* W2_p   = W1_p   + 256 * 1024;            // 1024*256

    const int nEl = LEN * CDIM;
    const dim3 blk(256);
    const dim3 gEl((nEl + 255) / 256);
    const dim3 gWaveTok((LEN + 7) / 8);                // 8 waves / block
    const dim3 gWaveTH((LEN * HEADS + 7) / 8);
    const dim3 gSoft((LEN * HEADS + 255) / 256);
    const int MB = (LEN + 255) / 256;                  // 74 row-blocks (32 rows/wave)

    auto packGrid = [](int N, int K) {
        return dim3(((N / 16) * (K / 32) * 32 + 255) / 256);
    };

    // h = x (f32 + f16 shadow) ; p = pos + level_embed
    copy_cvt_kernel<<<gEl, blk, 0, stream>>>(x, hbuf, h16);
    pos_level_kernel<<<gEl, blk, 0, stream>>>(pos, le, pbuf);

    for (int i = 0; i < NL; ++i) {
        const float* Wv_i   = Wv   + (size_t)i * CDIM * CDIM;
        const float* bv_i   = bv   + (size_t)i * CDIM;
        const float* Woff_i = Woff + (size_t)i * CDIM * (HEADS * NLVL * NPTS * 3);
        const float* boff_i = boff + (size_t)i * (HEADS * NLVL * NPTS * 3);
        const float* Wa_i   = Wa   + (size_t)i * CDIM * (HEADS * NLVL * NPTS);
        const float* ba_i   = ba   + (size_t)i * (HEADS * NLVL * NPTS);
        const float* Wout_i = Wout + (size_t)i * CDIM * CDIM;
        const float* bout_i = bout + (size_t)i * CDIM;
        const float* W1_i   = W1   + (size_t)i * CDIM * FF;
        const float* b1_i   = b1   + (size_t)i * FF;
        const float* W2_i   = W2   + (size_t)i * FF * CDIM;
        const float* b2_i   = b2   + (size_t)i * CDIM;

        // pack this layer's weights into f16 WMMA fragment order
        pack_b_kernel<<<packGrid(256,  256), blk, 0, stream>>>(Wv_i,   Wv_p,   256,  256);
        pack_b_kernel<<<packGrid(384,  256), blk, 0, stream>>>(Woff_i, Woff_p, 384,  256);
        pack_b_kernel<<<packGrid(128,  256), blk, 0, stream>>>(Wa_i,   Wa_p,   128,  256);
        pack_b_kernel<<<packGrid(256,  256), blk, 0, stream>>>(Wout_i, Wout_p, 256,  256);
        pack_b_kernel<<<packGrid(1024, 256), blk, 0, stream>>>(W1_i,   W1_p,   1024, 256);
        pack_b_kernel<<<packGrid(256, 1024), blk, 0, stream>>>(W2_i,   W2_p,   256, 1024);

        // q16 = (f16)(h + p)
        add_f16_kernel<<<gEl, blk, 0, stream>>>(hbuf, pbuf, q16, nEl);
        // value = h @ Wv + bv            [LEN,256] f32
        gemm_wmma_kernel<256, false, false><<<dim3(256 / 64, MB), blk, 0, stream>>>(
            h16, Wv_p, bv_i, vbuf, LEN, 256);
        // off = q @ Woff + boff          [LEN,384] f32
        gemm_wmma_kernel<256, false, false><<<dim3(384 / 64, MB), blk, 0, stream>>>(
            q16, Woff_p, boff_i, obuf, LEN, 384);
        // logits = q @ Wa + ba           [LEN,128] f32, softmax over 16
        gemm_wmma_kernel<256, false, false><<<dim3(128 / 64, MB), blk, 0, stream>>>(
            q16, Wa_p, ba_i, abuf, LEN, 128);
        softmax16_kernel<<<gSoft, blk, 0, stream>>>(abuf, LEN * HEADS);
        // trilinear deformable sampling -> at16 [LEN,256] f16
        deform_sample_kernel<<<gWaveTH, blk, 0, stream>>>(vbuf, obuf, abuf, at16);
        // tmp = attnout @ Wout + bout ; h = LN(h + tmp) (+ f16 shadow)
        gemm_wmma_kernel<256, false, false><<<dim3(256 / 64, MB), blk, 0, stream>>>(
            at16, Wout_p, bout_i, tmp, LEN, 256);
        add_ln_kernel<<<gWaveTok, blk, 0, stream>>>(
            hbuf, tmp, ln1s + (size_t)i * CDIM, ln1b + (size_t)i * CDIM, hbuf, h16);
        // ff16 = (f16)relu(h @ W1 + b1)  [LEN,1024]
        gemm_wmma_kernel<256, true, true><<<dim3(FF / 64, MB), blk, 0, stream>>>(
            h16, W1_p, b1_i, ff16, LEN, FF);
        // tmp = ff @ W2 + b2 ; h = LN(h + tmp)  (last layer -> d_out, no shadow)
        gemm_wmma_kernel<1024, false, false><<<dim3(256 / 64, MB), blk, 0, stream>>>(
            ff16, W2_p, b2_i, tmp, LEN, 256);
        const bool last = (i == NL - 1);
        add_ln_kernel<<<gWaveTok, blk, 0, stream>>>(
            hbuf, tmp, ln2s + (size_t)i * CDIM, ln2b + (size_t)i * CDIM,
            last ? out : hbuf, last ? (_Float16*)nullptr : h16);
    }
}
